// ResGCN_15195594293517
// MI455X (gfx1250) — matrix operand order repaired and checked by reading, not compiled, not run on hardware
//
#include <hip/hip_runtime.h>
#include <hip/hip_bf16.h>

// ---------------------------------------------------------------------------
// ResGCN on MI455X (gfx1250, wave32):
//   h   = relu(SpMM(x@W1)) + (x@Wres + bres)
//   out = relu(SpMM(h@W2)) + h
// Dense GEMMs: v_wmma_f32_16x16x32_bf16, fp32 accumulate.
//   - each wave computes a 16x64 output strip (4 WMMAs per K-step, A reused)
//   - B staged in LDS pre-swizzled into WMMA fragment order: one lane's
//     16-element B fragment is a single contiguous 32B chunk -> 2x ds_load_b128
//   - A loaded with b128 vector loads, converted fp32->bf16 in-register
// SpMM: edge-parallel gather + non-returning global_atomic_add_f32 scatter.
//   Feature buffers are 25.6MB -> fully L2-resident (192MB L2), so the random
//   gather/scatter is L2-bound, not HBM-bound.
// ---------------------------------------------------------------------------

typedef __attribute__((ext_vector_type(16))) __bf16 v16bf;
typedef __attribute__((ext_vector_type(8)))  float  v8f;

#define NNODES 50000
#define NEDGES 800000
#define DHID   128
#define KMAX   128   // max K of any GEMM here (layer 2)

// ---------------------------------------------------------------------------
// C[M,N] = A[M,K] @ B[K,N] (+ bias). Row-major fp32 in/out.
// Grid: x = ceil((M/16)/8) blocks of 8 waves (one 16-row tile per wave),
//       y = N/64 (64-column slice per block).
//
// LDS layout (fragment-order swizzle): chunk index (kk>>4)*64 + col holds the
// 16 bf16 values B[(kk>>4)*16 + e][nbase+col], e=0..15, as one v16bf.
// B-fragment for WMMA: lane group g=lane>>4 needs K = k0 + 16g + e  ->
// chunk ((k0>>4)+g)*64 + j*16 + (lane&15). Single 32B aligned LDS read.
// ---------------------------------------------------------------------------
__global__ void gemm_bf16_wmma(const float* __restrict__ A,
                               const float* __restrict__ B,
                               const float* __restrict__ bias,
                               float* __restrict__ C,
                               int M, int N, int K)
{
    __shared__ v16bf Bs[(KMAX >> 4) * 64];    // 512 chunks * 32B = 16KB max

    const int lane  = threadIdx.x & 31;
    const int wib   = threadIdx.x >> 5;       // wave in block (0..7)
    const int nbase = blockIdx.y * 64;        // first of our 64 columns

    // ---- stage B slice into LDS in fragment order (all threads) ----
    const int chunks = (K >> 4) * 64;
    for (int ch = (int)threadIdx.x; ch < chunks; ch += blockDim.x) {
        const int cc = ch & 63;               // column within slice
        const int kb = (ch >> 6) << 4;        // starting K of this chunk
        v16bf tmp;
#pragma unroll
        for (int e = 0; e < 16; ++e) {
            tmp[e] = (__bf16)B[(size_t)(kb + e) * N + nbase + cc];
        }
        Bs[ch] = tmp;
    }
    __syncthreads();

    const int tm = blockIdx.x * 8 + wib;      // 16-row tile index
    if (tm * 16 >= M) return;                 // wave-uniform: EXEC all-1s kept

    // A fragment addressing (16-bit A 16x32, contiguous K order shared with B, not ISA table order):
    //   lane<16: row=lane, K = {0..15};  lane>=16: K = {16..31}
    const int rowA  = lane & 15;
    const int khalf = (lane >> 4) << 4;       // 0 or 16
    const int ncol  = lane & 15;
    const int g     = lane >> 4;              // 0 or 1 (B K-group)

    v8f acc[4] = {{}, {}, {}, {}};

    for (int k0 = 0; k0 < K; k0 += 32) {
        // ---- A: four contiguous float4 loads + cvt to bf16 ----
        const float4* arow4 =
            (const float4*)(A + (size_t)(tm * 16 + rowA) * K + k0);
        const int q = khalf >> 2;             // 0 or 4
        const float4 a0 = arow4[q];
        const float4 a1 = arow4[q + 1];
        const float4 a2 = arow4[q + 2];
        const float4 a3 = arow4[q + 3];
        v16bf a;
        a[0]  = (__bf16)a0.x; a[1]  = (__bf16)a0.y;
        a[2]  = (__bf16)a0.z; a[3]  = (__bf16)a0.w;
        a[4]  = (__bf16)a1.x; a[5]  = (__bf16)a1.y;
        a[6]  = (__bf16)a1.z; a[7]  = (__bf16)a1.w;
        a[8]  = (__bf16)a2.x; a[9]  = (__bf16)a2.y;
        a[10] = (__bf16)a2.z; a[11] = (__bf16)a2.w;
        a[12] = (__bf16)a3.x; a[13] = (__bf16)a3.y;
        a[14] = (__bf16)a3.z; a[15] = (__bf16)a3.w;

        // ---- 4 column-tiles: one 32B LDS fragment read + WMMA each ----
        const int cbase = ((k0 >> 4) + g) * 64 + ncol;
#pragma unroll
        for (int j = 0; j < 4; ++j) {
            const v16bf b = Bs[cbase + j * 16];
            acc[j] = __builtin_amdgcn_wmma_f32_16x16x32_bf16(
                         false, a, false, b, (short)0, acc[j], false, false);
        }
    }

    // ---- store: D layout row=(lane>>4)*8+v, col=lane&15 ----
    const int ccol = lane & 15;
    const int crow = (lane >> 4) << 3;
#pragma unroll
    for (int j = 0; j < 4; ++j) {
        const float bv = bias ? bias[nbase + j * 16 + ccol] : 0.0f;
        float* cptr = C + (size_t)(tm * 16 + crow) * N + nbase + j * 16 + ccol;
#pragma unroll
        for (int v = 0; v < 8; ++v) {
            cptr[(size_t)v * N] = acc[j][v] + bv;
        }
    }
}

// ---------------------------------------------------------------------------
// dst[row[e], :] += edge_w[e] * src[col[e], :]   (D = 128)
// 32 lanes per edge, float4 per lane -> one full 512B row per edge.
// ---------------------------------------------------------------------------
__global__ void spmm_scatter(const int*   __restrict__ rowi,
                             const int*   __restrict__ coli,
                             const float* __restrict__ ew,
                             const float* __restrict__ src,
                             float*       __restrict__ dst)
{
    const int tid  = blockIdx.x * blockDim.x + threadIdx.x;
    const int e    = tid >> 5;
    if (e >= NEDGES) return;
    const int lane = tid & 31;

    const int   r = rowi[e];
    const int   c = coli[e];
    const float w = ew[e];

    const float4 v = ((const float4*)(src + (size_t)c * DHID))[lane];
    float* d = dst + (size_t)r * DHID + (size_t)lane * 4;
    atomicAdd(d + 0, v.x * w);
    atomicAdd(d + 1, v.y * w);
    atomicAdd(d + 2, v.z * w);
    atomicAdd(d + 3, v.w * w);
}

__global__ void fill_zero4(float4* __restrict__ p, int n4)
{
    const int i = blockIdx.x * blockDim.x + threadIdx.x;
    if (i < n4) p[i] = make_float4(0.f, 0.f, 0.f, 0.f);
}

// out = relu(agg) + res, vectorized float4
__global__ void relu_add4(const float4* __restrict__ agg,
                          const float4* __restrict__ res,
                          float4*       __restrict__ out, int n4)
{
    const int i = blockIdx.x * blockDim.x + threadIdx.x;
    if (i >= n4) return;
    const float4 a = agg[i];
    const float4 r = res[i];
    float4 o;
    o.x = fmaxf(a.x, 0.f) + r.x;
    o.y = fmaxf(a.y, 0.f) + r.y;
    o.z = fmaxf(a.z, 0.f) + r.z;
    o.w = fmaxf(a.w, 0.f) + r.w;
    out[i] = o;
}

extern "C" void kernel_launch(void* const* d_in, const int* in_sizes, int n_in,
                              void* d_out, int out_size, void* d_ws, size_t ws_size,
                              hipStream_t stream)
{
    const float* x    = (const float*)d_in[0];   // [50000, 64]
    const float* ew   = (const float*)d_in[1];   // [800000]
    const float* W1   = (const float*)d_in[2];   // [64, 128]
    const float* W2   = (const float*)d_in[3];   // [128, 128]
    const float* Wres = (const float*)d_in[4];   // [64, 128]
    const float* bres = (const float*)d_in[5];   // [128]
    const int*   row  = (const int*)d_in[6];     // [800000]
    const int*   col  = (const int*)d_in[7];     // [800000]
    float*       out  = (float*)d_out;           // [50000, 128]

    // Workspace: 4 buffers of 50000*128 fp32 = 102.4MB
    const size_t NM  = (size_t)NNODES * DHID;
    float* xw1 = (float*)d_ws;        // x@W1, later reused for h@W2
    float* res = xw1 + NM;            // x@Wres + bres
    float* agg = res + NM;            // SpMM accumulator
    float* h   = agg + NM;            // layer-1 output

    const dim3 blk(256);
    // GEMM grid: x covers 3125 row-tiles at 8 waves/block; y = 128/64 = 2
    const dim3 ggrid((NNODES / 16 + 7) / 8, DHID / 64);        // (391, 2)
    const int  n4 = (int)(NM / 4);
    const dim3 fgrid((n4 + 255) / 256);
    const dim3 sgrid((NEDGES * 32 + 255) / 256);               // 100000 blocks

    // ---- Layer 1 ----
    gemm_bf16_wmma<<<ggrid, blk, 0, stream>>>(x, W1,   nullptr, xw1, NNODES, DHID, 64);
    gemm_bf16_wmma<<<ggrid, blk, 0, stream>>>(x, Wres, bres,    res, NNODES, DHID, 64);
    fill_zero4<<<fgrid, blk, 0, stream>>>((float4*)agg, n4);
    spmm_scatter<<<sgrid, blk, 0, stream>>>(row, col, ew, xw1, agg);
    relu_add4<<<fgrid, blk, 0, stream>>>((const float4*)agg, (const float4*)res,
                                         (float4*)h, n4);

    // ---- Layer 2 ----
    gemm_bf16_wmma<<<ggrid, blk, 0, stream>>>(h, W2, nullptr, xw1, NNODES, DHID, 128);
    fill_zero4<<<fgrid, blk, 0, stream>>>((float4*)agg, n4);
    spmm_scatter<<<sgrid, blk, 0, stream>>>(row, col, ew, xw1, agg);
    relu_add4<<<fgrid, blk, 0, stream>>>((const float4*)agg, (const float4*)h,
                                         (float4*)out, n4);
}
